// MultiheadAttention_27925877358836
// MI455X (gfx1250) — compile-verified, matrix operand order checked by reading, not run on hardware
//
#include <hip/hip_runtime.h>
#include <hip/hip_bf16.h>

// MHA forward for B=4, S=2048, D=1024, H=16, DK=64 on gfx1250 (wave32, WMMA).
// Pipeline: f32->f16 convert -> 3 projection GEMMs (WMMA, A tile via
// async-to-LDS, W tile via TDM tensor_load_to_lds) -> flash attention
// (WMMA + online softmax, async K tiles) -> output projection GEMM (f32 out).

typedef _Float16 h16;
typedef __attribute__((ext_vector_type(4)))  _Float16 v4h;
typedef __attribute__((ext_vector_type(8)))  _Float16 v8h;
typedef __attribute__((ext_vector_type(16))) _Float16 v16h;
typedef __attribute__((ext_vector_type(8)))  float    v8f;
typedef __attribute__((ext_vector_type(4)))  unsigned v4u;
typedef __attribute__((ext_vector_type(8)))  unsigned v8u;

#define DEV static __device__ __forceinline__

static constexpr int Bc  = 4;
static constexpr int Sc  = 2048;
static constexpr int Dc  = 1024;
static constexpr int Hc  = 16;
static constexpr int DKc = 64;
static constexpr int Mc  = Bc * Sc;      // 8192 rows for all GEMMs

// ---------------- CDNA5 async / tensor data-mover paths ---------------------
#if defined(__gfx1250__)
#  define ASYNC_PATH 1
#else
#  define ASYNC_PATH 0
#endif

DEV void cp16_g2l(const h16* g, h16* l) {
#if ASYNC_PATH
  // GLOBAL_LOAD_ASYNC_TO_LDS_B128: per-lane copy of 16B global -> LDS.
  // VDST = LDS byte address (generic LDS ptr truncated to 32 bits, ISA 10.2),
  // VADDR = 64-bit global address, GV mode (saddr = off). Tracked on ASYNCcnt.
  const unsigned lds_addr = (unsigned)(unsigned long long)l;
  asm volatile("global_load_async_to_lds_b128 %0, %1, off"
               :
               : "v"(lds_addr), "v"(g)
               : "memory");
#else
  *(v8h*)l = *(const v8h*)g;
#endif
}

template <int N> DEV void wait_async() {
#if ASYNC_PATH
#  if __has_builtin(__builtin_amdgcn_s_wait_asynccnt)
  __builtin_amdgcn_s_wait_asynccnt(N);
#  else
  asm volatile("s_wait_asynccnt %0" ::"i"(N) : "memory");
#  endif
#endif
}

template <int N> DEV void wait_tensor() {
#if ASYNC_PATH
#  if __has_builtin(__builtin_amdgcn_s_wait_tensorcnt)
  __builtin_amdgcn_s_wait_tensorcnt(N);
#  else
  asm volatile("s_wait_tensorcnt %0" ::"i"(N) : "memory");
#  endif
#endif
}

#if ASYNC_PATH
// TDM: load a 128-row x 32-col f16 tile (row stride 1024 elems) from global
// into LDS with 8-half row padding (LDS pitch 40 halves). One instruction per
// wave; tracked on TENSORcnt. D# layout per CDNA5 ISA ch.8.
DEV void tdm_load_w_tile(const h16* gtile, unsigned lds_byte_addr) {
  const unsigned long long ga = (unsigned long long)gtile;
  v4u g0;
  g0[0] = 1u;                                   // count=1, user descriptor
  g0[1] = lds_byte_addr;                        // lds_addr
  g0[2] = (unsigned)ga;                         // global_addr[31:0]
  g0[3] = (unsigned)((ga >> 32) & 0x01FFFFFFu)  // global_addr[56:32]
          | (2u << 30);                         // type = 2 ("image")
  v8u g1;
  g1[0] = (1u << 16)     // data_size = 1 -> 2 bytes
        | (1u << 20)     // pad_enable
        | (3u << 22)     // pad_interval: code 3 -> every 16 DWORDs (one row)
        | (3u << 25);    // pad_amount:   code 3 -> 4 DWORDs (8 halves)
  g1[1] = (1024u << 16); // tensor_dim0 = 1024 (bits 79:48, low half here)
  g1[2] = (1024u << 16); // tensor_dim0 hi = 0 | tensor_dim1 = 1024 (low half)
  g1[3] = (32u << 16);   // tensor_dim1 hi = 0 | tile_dim0 = 32
  g1[4] = 128u;          // tile_dim1 = 128 | tile_dim2 = 0
  g1[5] = 1024u;         // tensor_dim0_stride[31:0] = 1024 elems
  g1[6] = 0u;            // stride hi | tensor_dim1_stride lo
  g1[7] = 0u;
  asm volatile("tensor_load_to_lds %0, %1" ::"s"(g0), "s"(g1) : "memory");
}
#endif

// ---------------------------------------------------------------------------
DEV v8f wmma16(v16h a, v16h b, v8f c) {
  // D = A(16x32 f16) * B(32x16 f16) + C(16x16 f32)
  return __builtin_amdgcn_wmma_f32_16x16x32_f16(false, a, false, b,
                                                (short)0, c, false, false);
}

DEV v16h ldfrag(const h16* p, int ks) {
  // A/B fragment: halves {ks..ks+7} and {ks+16..ks+23} of a 32-deep K window.
  v8h lo = *(const v8h*)(p + ks);
  v8h hi = *(const v8h*)(p + ks + 16);
  return __builtin_shufflevector(lo, hi, 0,1,2,3,4,5,6,7,8,9,10,11,12,13,14,15);
}

template <int MASK>
DEV float swz(float x) {
  // ds_swizzle group-of-32 XOR pattern: and=0x1f, or=0, xor=MASK
  int r = __builtin_amdgcn_ds_swizzle(__builtin_bit_cast(int, x),
                                      (MASK << 10) | 0x1F);
  return __builtin_bit_cast(float, r);
}

DEV v8f vzero() {
  v8f z;
#pragma unroll
  for (int i = 0; i < 8; ++i) z[i] = 0.0f;
  return z;
}

// ---------------------------------------------------------------- convert ---
__global__ __launch_bounds__(256)
void f32_to_f16_vec(const float* __restrict__ src, h16* __restrict__ dst, int n4) {
  int i = blockIdx.x * 256 + threadIdx.x;
  if (i < n4) {
    const float4 f = ((const float4*)src)[i];
    v4h o;
    o[0] = (h16)f.x; o[1] = (h16)f.y; o[2] = (h16)f.z; o[3] = (h16)f.w;
    ((v4h*)dst)[i] = o;
  }
}

// ------------------------------------------------------------------- GEMM ---
// OUT[M,N] = A[M,K] (f16) @ W[N,K]^T (f16) + bias[N]
// mode 0: write f16 head-split layout [B,H,S,DK]   (QKV projections)
// mode 1: write f32 row-major [M,N]                (output projection)
__global__ __launch_bounds__(256)
void gemm128(const h16* __restrict__ A, const h16* __restrict__ W,
             const float* __restrict__ bias,
             h16* __restrict__ outh, float* __restrict__ outf, int mode) {
  constexpr int BM = 128, BN = 128, BK = 32, PITCH = 40;
  __shared__ __align__(16) h16 As[2][BM][PITCH];
  __shared__ __align__(16) h16 Bs[2][BN][PITCH];

  const int bn0  = blockIdx.x * BN;
  const int bm0  = blockIdx.y * BM;
  const int tid  = threadIdx.x;
  const int wave = tid >> 5;
  const int lane = tid & 31;
  const int wm   = (wave & 3) * 32;   // wave tile: 32 rows x 64 cols
  const int wn   = (wave >> 2) * 64;
  const int rl   = lane & 15;
  const int ks   = (lane >> 4) * 8;

  v8f acc[2][4];
#pragma unroll
  for (int mt = 0; mt < 2; ++mt)
#pragma unroll
    for (int nt = 0; nt < 4; ++nt) acc[mt][nt] = vzero();

  const int lrow = tid >> 1;           // 0..127
  const int lcb  = (tid & 1) * 16;     // 0 / 16
  const h16* aG = A + (size_t)(bm0 + lrow) * Dc + lcb;

  // A tile: 2 async ops per thread per tile (2x16B)
  auto issueA = [&](int k0, int buf) {
    cp16_g2l(aG + k0,     &As[buf][lrow][lcb]);
    cp16_g2l(aG + k0 + 8, &As[buf][lrow][lcb + 8]);
  };
#if ASYNC_PATH
  // W tile: one TDM descriptor per tile, issued by wave 0 only.
  auto issueB = [&](int k0, int buf) {
    if (wave == 0)
      tdm_load_w_tile(W + (size_t)bn0 * Dc + k0,
                      (unsigned)(unsigned long long)&Bs[buf][0][0]);
  };
#else
  const h16* bG = W + (size_t)(bn0 + lrow) * Dc + lcb;
  auto issueB = [&](int k0, int buf) {
    cp16_g2l(bG + k0,     &Bs[buf][lrow][lcb]);
    cp16_g2l(bG + k0 + 8, &Bs[buf][lrow][lcb + 8]);
  };
#endif

  issueA(0, 0);
  issueB(0, 0);
  for (int k0 = 0, it = 0; k0 < Dc; k0 += BK, ++it) {
    const int buf = it & 1;
    if (k0 + BK < Dc) {
      issueA(k0 + BK, buf ^ 1);
      issueB(k0 + BK, buf ^ 1);
      wait_async<2>();                      // 2 oldest (current A tile) done
#if ASYNC_PATH
      if (wave == 0) wait_tensor<1>();      // current W tile done; next in flight
#endif
    } else {
      wait_async<0>();
#if ASYNC_PATH
      if (wave == 0) wait_tensor<0>();
#endif
    }
    __syncthreads();

    v16h af[2];
#pragma unroll
    for (int mt = 0; mt < 2; ++mt)
      af[mt] = ldfrag(&As[buf][wm + mt * 16 + rl][0], ks);
#pragma unroll
    for (int nt = 0; nt < 4; ++nt) {
      v16h bf = ldfrag(&Bs[buf][wn + nt * 16 + rl][0], ks);
      acc[0][nt] = wmma16(af[0], bf, acc[0][nt]);
      acc[1][nt] = wmma16(af[1], bf, acc[1][nt]);
    }
    __syncthreads();
  }

#pragma unroll
  for (int nt = 0; nt < 4; ++nt) {
    const int n  = bn0 + wn + nt * 16 + rl;
    const float bv = bias[n];
#pragma unroll
    for (int mt = 0; mt < 2; ++mt) {
#pragma unroll
      for (int r = 0; r < 8; ++r) {
        const int m = bm0 + wm + mt * 16 + ks + r;   // C layout: M = (l>>4)*8 + r
        const float val = acc[mt][nt][r] + bv;
        if (mode == 0) {
          const int bb = m >> 11, ss = m & (Sc - 1);
          const int hh = n >> 6,  dk = n & (DKc - 1);
          outh[(((size_t)(bb * Hc + hh)) * Sc + ss) * DKc + dk] = (h16)val;
        } else {
          outf[(size_t)m * Dc + n] = val;
        }
      }
    }
  }
}

// -------------------------------------------------------------- attention ---
// Q/K/V in f16 [B*H][S][DK]; Z out f16 merged-head [B][S][D].
__global__ __launch_bounds__(256)
void flash_attn(const h16* __restrict__ Qh, const h16* __restrict__ Kh,
                const h16* __restrict__ Vh, h16* __restrict__ Zh) {
  __shared__ __align__(16) h16 Ksm[2][64][72];    // [kv][dk], double-buffered
  __shared__ __align__(16) h16 Vtm[2][64][72];    // [dk][kv] (transposed)
  __shared__ __align__(16) h16 Psm[8][16][72];    // per-wave P staging

  const int bh   = blockIdx.y;              // b*H + h
  const int b    = bh >> 4;
  const int h    = bh & (Hc - 1);
  const int tid  = threadIdx.x;
  const int wave = tid >> 5;
  const int lane = tid & 31;
  const int rl   = lane & 15;
  const int ks   = (lane >> 4) * 8;
  const size_t base = (size_t)bh * Sc * DKc;
  const int q0   = blockIdx.x * 128 + wave * 16;  // this wave's 16 query rows

  // Q fragments (held in registers, pre-scaled by 1/sqrt(DK) = 0.125)
  v16h qf[2];
  {
    const h16* qrow = Qh + base + (size_t)(q0 + rl) * DKc;
    const h16 sc = (h16)0.125f;
    v8h a = *(const v8h*)(qrow + ks)           * sc;
    v8h c = *(const v8h*)(qrow + ks + 16)      * sc;
    v8h d = *(const v8h*)(qrow + 32 + ks)      * sc;
    v8h e = *(const v8h*)(qrow + 32 + ks + 16) * sc;
    qf[0] = __builtin_shufflevector(a, c, 0,1,2,3,4,5,6,7,8,9,10,11,12,13,14,15);
    qf[1] = __builtin_shufflevector(d, e, 0,1,2,3,4,5,6,7,8,9,10,11,12,13,14,15);
  }

  float rm[8], rs[8];
  v8f zacc[4];
#pragma unroll
  for (int r = 0; r < 8; ++r) { rm[r] = -1e30f; rs[r] = 0.0f; }
#pragma unroll
  for (int nt = 0; nt < 4; ++nt) zacc[nt] = vzero();

  const int crow = tid >> 2;          // 0..63
  const int ccb  = (tid & 3) * 16;    // 0/16/32/48
  const h16* kG = Kh + base + (size_t)crow * DKc + ccb;
  const h16* vG = Vh + base + (size_t)crow * DKc + ccb;

  constexpr int NT = Sc / 64;         // 32 kv tiles

  // K tile: async (2 ops/thread). V tile: sync regs + transpose scatter.
  auto issueK = [&](int it, int buf) {
    const size_t off = (size_t)it * 64 * DKc;
    cp16_g2l(kG + off,     &Ksm[buf][crow][ccb]);
    cp16_g2l(kG + off + 8, &Ksm[buf][crow][ccb + 8]);
  };
  auto scatterV = [&](v8h v0, v8h v1, int buf) {
#pragma unroll
    for (int i = 0; i < 8; ++i) {
      Vtm[buf][ccb + i][crow]     = v0[i];
      Vtm[buf][ccb + 8 + i][crow] = v1[i];
    }
  };

  // prologue: tile 0
  issueK(0, 0);
  {
    v8h v0 = *(const v8h*)(vG);
    v8h v1 = *(const v8h*)(vG + 8);
    scatterV(v0, v1, 0);
  }

  for (int it = 0; it < NT; ++it) {
    const int buf = it & 1;
    v8h nv0, nv1;
    if (it + 1 < NT) {
      issueK(it + 1, buf ^ 1);
      const size_t off = (size_t)(it + 1) * 64 * DKc;
      nv0 = *(const v8h*)(vG + off);
      nv1 = *(const v8h*)(vG + off + 8);
      wait_async<2>();   // 2 oldest (current K tile) complete
    } else {
      wait_async<0>();
    }
    __syncthreads();

    // scores: S-tile = Q(16x64) @ K^T -> 4 tiles of 16x16
    v8f sc4[4];
#pragma unroll
    for (int nt = 0; nt < 4; ++nt) {
      sc4[nt] = vzero();
      const h16* bp = &Ksm[buf][nt * 16 + rl][0];
      sc4[nt] = wmma16(qf[0], ldfrag(bp, ks),      sc4[nt]);
      sc4[nt] = wmma16(qf[1], ldfrag(bp + 32, ks), sc4[nt]);
    }

    // online softmax per row (C layout: row = ks + r, col = nt*16 + rl)
#pragma unroll
    for (int r = 0; r < 8; ++r) {
      const float s0 = sc4[0][r], s1 = sc4[1][r], s2 = sc4[2][r], s3 = sc4[3][r];
      float tm = fmaxf(fmaxf(s0, s1), fmaxf(s2, s3));
      tm = fmaxf(tm, swz<1>(tm)); tm = fmaxf(tm, swz<2>(tm));
      tm = fmaxf(tm, swz<4>(tm)); tm = fmaxf(tm, swz<8>(tm));
      const float mnew = fmaxf(rm[r], tm);
      const float c0 = 1.442695041f;
      const float p0 = exp2f((s0 - mnew) * c0);
      const float p1 = exp2f((s1 - mnew) * c0);
      const float p2 = exp2f((s2 - mnew) * c0);
      const float p3 = exp2f((s3 - mnew) * c0);
      float lsum = p0 + p1 + p2 + p3;
      lsum += swz<1>(lsum); lsum += swz<2>(lsum);
      lsum += swz<4>(lsum); lsum += swz<8>(lsum);
      const float corr = exp2f((rm[r] - mnew) * c0);
      rs[r] = rs[r] * corr + lsum;
      rm[r] = mnew;
      zacc[0][r] *= corr; zacc[1][r] *= corr;
      zacc[2][r] *= corr; zacc[3][r] *= corr;
      h16* pr = &Psm[wave][ks + r][0];
      pr[rl]      = (h16)p0;
      pr[16 + rl] = (h16)p1;
      pr[32 + rl] = (h16)p2;
      pr[48 + rl] = (h16)p3;
    }

    // Z += P(16x64) @ V(64x64): A frags from Psm, B frags from Vtm[dk][kv]
    {
      const h16* pp = &Psm[wave][rl][0];
      const v16h pf0 = ldfrag(pp, ks);
      const v16h pf1 = ldfrag(pp + 32, ks);
#pragma unroll
      for (int nt = 0; nt < 4; ++nt) {
        const h16* vp = &Vtm[buf][nt * 16 + rl][0];
        zacc[nt] = wmma16(pf0, ldfrag(vp, ks),      zacc[nt]);
        zacc[nt] = wmma16(pf1, ldfrag(vp + 32, ks), zacc[nt]);
      }
    }

    // stage next V tile while this tile's compute retires
    if (it + 1 < NT) scatterV(nv0, nv1, buf ^ 1);
    __syncthreads();
  }

  // normalize + store merged-head f16 [B][S][D]
#pragma unroll
  for (int r = 0; r < 8; ++r) {
    const float inv = 1.0f / rs[r];
    const int srow = q0 + ks + r;
    h16* zr = Zh + ((size_t)b * Sc + srow) * Dc + h * DKc;
    zr[rl]      = (h16)(zacc[0][r] * inv);
    zr[16 + rl] = (h16)(zacc[1][r] * inv);
    zr[32 + rl] = (h16)(zacc[2][r] * inv);
    zr[48 + rl] = (h16)(zacc[3][r] * inv);
  }
}

// ------------------------------------------------------------------ launch ---
extern "C" void kernel_launch(void* const* d_in, const int* in_sizes, int n_in,
                              void* d_out, int out_size, void* d_ws, size_t ws_size,
                              hipStream_t stream) {
  const float* q  = (const float*)d_in[0];
  const float* k  = (const float*)d_in[1];
  const float* v  = (const float*)d_in[2];
  const float* Wq = (const float*)d_in[3];  const float* bq = (const float*)d_in[4];
  const float* Wk = (const float*)d_in[5];  const float* bk = (const float*)d_in[6];
  const float* Wv = (const float*)d_in[7];  const float* bv = (const float*)d_in[8];
  const float* Wo = (const float*)d_in[9];  const float* bo = (const float*)d_in[10];
  float* out = (float*)d_out;

  const size_t MK = (size_t)Mc * Dc;   // 8M elems
  const size_t DD = (size_t)Dc * Dc;   // 1M elems
  h16* xq  = (h16*)d_ws;
  h16* xk  = xq  + MK;
  h16* xv  = xk  + MK;
  h16* wqh = xv  + MK;
  h16* wkh = wqh + DD;
  h16* wvh = wkh + DD;
  h16* woh = wvh + DD;
  h16* qh  = woh + DD;
  h16* kh  = qh  + MK;
  h16* vh  = kh  + MK;
  h16* zh  = vh  + MK;

  const int mk4 = (int)(MK / 4), dd4 = (int)(DD / 4);
  f32_to_f16_vec<<<mk4 / 256, 256, 0, stream>>>(q,  xq,  mk4);
  f32_to_f16_vec<<<mk4 / 256, 256, 0, stream>>>(k,  xk,  mk4);
  f32_to_f16_vec<<<mk4 / 256, 256, 0, stream>>>(v,  xv,  mk4);
  f32_to_f16_vec<<<dd4 / 256, 256, 0, stream>>>(Wq, wqh, dd4);
  f32_to_f16_vec<<<dd4 / 256, 256, 0, stream>>>(Wk, wkh, dd4);
  f32_to_f16_vec<<<dd4 / 256, 256, 0, stream>>>(Wv, wvh, dd4);
  f32_to_f16_vec<<<dd4 / 256, 256, 0, stream>>>(Wo, woh, dd4);

  dim3 gg(Dc / 128, Mc / 128);  // 8 x 64
  gemm128<<<gg, 256, 0, stream>>>(xq, wqh, bq, qh, nullptr, 0);
  gemm128<<<gg, 256, 0, stream>>>(xk, wkh, bk, kh, nullptr, 0);
  gemm128<<<gg, 256, 0, stream>>>(xv, wvh, bv, vh, nullptr, 0);

  flash_attn<<<dim3(Sc / 128, Bc * Hc), 256, 0, stream>>>(qh, kh, vh, zh);

  gemm128<<<gg, 256, 0, stream>>>(zh, woh, bo, nullptr, out, 1);
}
